// AnalyticFlow_25632364822572
// MI455X (gfx1250) — matrix operand order repaired and checked by reading, not compiled, not run on hardware
//
#include <hip/hip_runtime.h>
#include <hip/hip_bf16.h>

// ---------------------------------------------------------------------------
// Problem constants (from reference): B=256, P=50000, D=C*H*W=3072
// ---------------------------------------------------------------------------
static constexpr int   kB    = 256;
static constexpr int   kP    = 50000;
static constexpr int   kD    = 3072;
static constexpr int   kPpad = 50048;          // 391*128 == 1564*32
static constexpr float kT    = 999.0f;

typedef __attribute__((ext_vector_type(16))) __bf16 v16bf;
typedef __attribute__((ext_vector_type(8)))  __bf16 v8bf;
typedef __attribute__((ext_vector_type(8)))  float  v8f;
typedef unsigned int u32x4 __attribute__((ext_vector_type(4)));
typedef int          i32x4 __attribute__((ext_vector_type(4)));
typedef int          i32x8 __attribute__((ext_vector_type(8)));

static constexpr int PF  = 36;   // f32 LDS pitch, 32-dword TDM tiles (+4 pad dwords)
static constexpr int PG  = 68;   // f32 LDS pitch, 64-dword TDM tiles (+4 pad dwords)
static constexpr int PBF = 40;   // bf16 LDS pitch for converted tiles
static constexpr int PBT = 40;   // bf16 LDS pitch for gemm2 transposed tile

union BF4 { __bf16 h[4]; uint2 u; };

__device__ __forceinline__ v8bf ld8(const __bf16* p) { return *(const v8bf*)p; }

__device__ __forceinline__ v16bf mk16(v8bf lo, v8bf hi) {
  v16bf r;
#pragma unroll
  for (int i = 0; i < 8; ++i) { r[i] = lo[i]; r[i + 8] = hi[i]; }
  return r;
}

// ---------------------------------------------------------------------------
// Tensor Data Mover: load a [tileH x tileW] f32 tile (row stride strideElems)
// from global into LDS at lds_byte_addr, LDS rows padded by 4 dwords
// (pad_interval code: 4 -> every 32 dwords, 5 -> every 64 dwords).
// D# layout per CDNA5 ISA ch.8; rows y >= rowsValid read as zero.
// ---------------------------------------------------------------------------
__device__ __forceinline__ void tdm_load_2d_f32(
    unsigned lds_byte_addr, const float* src, int rowsValid,
    int strideElems, int tileW, int tileH, int padIntervalCode) {
  if (rowsValid < 0) rowsValid = 0;
  unsigned long long ga = (unsigned long long)(uintptr_t)src;
  u32x4 g0;
  g0[0] = 1u;                                          // count=1 (user, no gather)
  g0[1] = lds_byte_addr;                               // lds_addr
  g0[2] = (unsigned)(ga & 0xFFFFFFFFu);                // global_addr[31:0]
  g0[3] = (unsigned)((ga >> 32) & 0x01FFFFFFu)         // global_addr[56:32]
        | (2u << 30);                                  // type = 2 ("image")
  i32x8 g1;
  g1[0] = (int)((2u << 16)                             // data_size = 4 bytes
              | (1u << 20)                             // pad_enable
              | ((unsigned)padIntervalCode << 22)      // pad interval
              | (3u << 25));                           // pad_amount: 4 dwords
  g1[1] = (int)(((unsigned)tileW & 0xFFFFu) << 16);    // tensor_dim0[15:0]
  g1[2] = (int)(((unsigned)rowsValid & 0xFFFFu) << 16);// tensor_dim1[15:0]
  g1[3] = (int)((((unsigned)rowsValid >> 16) & 0xFFFFu)// tensor_dim1[31:16]
              | (((unsigned)tileW) << 16));            // tile_dim0
  g1[4] = tileH;                                       // tile_dim1
  g1[5] = strideElems;                                 // tensor_dim0_stride[31:0]
  g1[6] = 0;
  g1[7] = 0;
  i32x4 z4 = {0, 0, 0, 0};                             // 2D: groups 2/3 unused
  i32x8 z8 = {0, 0, 0, 0, 0, 0, 0, 0};
  __builtin_amdgcn_tensor_load_to_lds(g0, g1, z4, z4, z8, 0);
}

// ---------------------------------------------------------------------------
// K1: per-batch scalars: tn, quad=|x|^2, inv_var, 1/(1-tn)
// ---------------------------------------------------------------------------
__global__ __launch_bounds__(256) void prep_b_kernel(
    const float* __restrict__ xt, const float* __restrict__ t,
    float* __restrict__ tn, float* __restrict__ quad,
    float* __restrict__ invvar, float* __restrict__ inv1mt) {
  int b = blockIdx.x;
  const float* row = xt + (size_t)b * kD;
  float s = 0.f;
  for (int i = threadIdx.x; i < kD; i += 256) { float v = row[i]; s += v * v; }
  __shared__ float red[256];
  red[threadIdx.x] = s; __syncthreads();
  for (int off = 128; off; off >>= 1) {
    if ((int)threadIdx.x < off) red[threadIdx.x] += red[threadIdx.x + off];
    __syncthreads();
  }
  if (threadIdx.x == 0) {
    float tnv = t[b] / kT;
    float om  = 1.0f - tnv;
    tn[b]     = tnv;
    quad[b]   = red[0];
    invvar[b] = 1.0f / (2.0f * om * om);
    inv1mt[b] = 1.0f / om;
  }
}

// ---------------------------------------------------------------------------
// K2: per-database-point squared norms
// ---------------------------------------------------------------------------
__global__ __launch_bounds__(256) void prep_p_kernel(
    const float* __restrict__ gt, float* __restrict__ sq) {
  int p = blockIdx.x;
  const float* row = gt + (size_t)p * kD;
  float s = 0.f;
  for (int i = threadIdx.x; i < kD; i += 256) { float v = row[i]; s += v * v; }
  __shared__ float red[256];
  red[threadIdx.x] = s; __syncthreads();
  for (int off = 128; off; off >>= 1) {
    if ((int)threadIdx.x < off) red[threadIdx.x] += red[threadIdx.x + off];
    __syncthreads();
  }
  if (threadIdx.x == 0) sq[p] = red[0];
}

// ---------------------------------------------------------------------------
// K3: logits GEMM.  S = x . gt^T via bf16 WMMA, fused logit epilogue.
// Tile: 128(M=B) x 128(N=P), K-step 32.  8 waves as 4(M) x 2(N).
// Pipeline per K-step:
//   wave0: issue TDM for step k+1 (f32, double-buffered), s_wait_tensorcnt
//   barrier; bulk convert f32 tile -> compact bf16 tiles (pitch 40)
//   barrier; fragments are pure bf16 ds_load_b128 -> v_wmma (no cvt in hot path)
// ---------------------------------------------------------------------------
__global__ __launch_bounds__(256) void gemm1_kernel(
    const float* __restrict__ x, const float* __restrict__ gt,
    const float* __restrict__ tn, const float* __restrict__ quad,
    const float* __restrict__ invvar, const float* __restrict__ sq,
    float* __restrict__ logits) {
  __shared__ float  Af32[2][128 * PF];
  __shared__ float  Bf32[2][128 * PF];
  __shared__ __bf16 Abf[128 * PBF];
  __shared__ __bf16 Bbf[128 * PBF];

  const int tid    = threadIdx.x;
  const int lane   = tid & 31;
  const int wave   = tid >> 5;
  const int lane15 = lane & 15;
  const int laneHi = lane >> 4;
  const int mWave  = wave & 3;      // 0..3 -> 32 M rows each
  const int nWave  = wave >> 2;     // 0..1 -> 64 N cols each
  const int mTile  = blockIdx.y * 128;
  const int nTile  = blockIdx.x * 128;
  const int rowsB  = kP - nTile;    // valid gt rows in this tile (>=80)

  const unsigned ldsA[2] = { (unsigned)(uintptr_t)&Af32[0][0],
                             (unsigned)(uintptr_t)&Af32[1][0] };
  const unsigned ldsB[2] = { (unsigned)(uintptr_t)&Bf32[0][0],
                             (unsigned)(uintptr_t)&Bf32[1][0] };

  v8f acc[2][4] = {};
  constexpr int nSteps = kD / 32;   // 96

  if (wave == 0) {
    tdm_load_2d_f32(ldsA[0], x  + (size_t)mTile * kD, 128,   kD, 32, 128, 4);
    tdm_load_2d_f32(ldsB[0], gt + (size_t)nTile * kD, rowsB, kD, 32, 128, 4);
  }

  for (int ks = 0; ks < nSteps; ++ks) {
    const int buf = ks & 1;
    if (wave == 0) {
      if (ks + 1 < nSteps) {
        const int k0n = (ks + 1) * 32;
        tdm_load_2d_f32(ldsA[buf ^ 1], x  + (size_t)mTile * kD + k0n, 128,   kD, 32, 128, 4);
        tdm_load_2d_f32(ldsB[buf ^ 1], gt + (size_t)nTile * kD + k0n, rowsB, kD, 32, 128, 4);
        __builtin_amdgcn_s_wait_tensorcnt(2);   // tiles for step ks are done
      } else {
        __builtin_amdgcn_s_wait_tensorcnt(0);
      }
    }
    __syncthreads();  // B1: f32 tile ks visible; prior compute done (bf16 reuse safe)

    // ---- bulk convert: f32 tiles -> compact bf16 tiles ----
    {
      const float* As = Af32[buf];
      const float* Bs = Bf32[buf];
#pragma unroll
      for (int i = 0; i < 4; ++i) {
        int idx = tid + i * 256;          // 0..1023 float4 slots
        int r   = idx >> 3;               // 0..127
        int c4  = idx & 7;                // 0..7
        float4 va = *(const float4*)(As + r * PF + c4 * 4);
        float4 vb = *(const float4*)(Bs + r * PF + c4 * 4);
        BF4 pa, pb;
        pa.h[0]=(__bf16)va.x; pa.h[1]=(__bf16)va.y; pa.h[2]=(__bf16)va.z; pa.h[3]=(__bf16)va.w;
        pb.h[0]=(__bf16)vb.x; pb.h[1]=(__bf16)vb.y; pb.h[2]=(__bf16)vb.z; pb.h[3]=(__bf16)vb.w;
        *(uint2*)&Abf[r * PBF + c4 * 4] = pa.u;
        *(uint2*)&Bbf[r * PBF + c4 * 4] = pb.u;
      }
    }
    __syncthreads();  // B2: bf16 tiles ready

    // ---- hot path: bf16 fragments + WMMA ----
    v16bf a[2];
#pragma unroll
    for (int i = 0; i < 2; ++i) {
      int m  = mWave * 32 + i * 16 + lane15;
      int kh = laneHi * 8;                      // lanes>=16: K 8..15 / 24..31
      a[i] = mk16(ld8(&Abf[m * PBF + kh]), ld8(&Abf[m * PBF + kh + 16]));
    }
#pragma unroll
    for (int j = 0; j < 4; ++j) {
      int p   = nWave * 64 + j * 16 + lane15;
      int ks2 = laneHi * 8;                     // lanes>=16: K 8..15 / 24..31
      v16bf bfr = mk16(ld8(&Bbf[p * PBF + ks2]), ld8(&Bbf[p * PBF + ks2 + 16]));
#pragma unroll
      for (int i = 0; i < 2; ++i) {
        acc[i][j] = __builtin_amdgcn_wmma_f32_16x16x32_bf16(
            false, a[i], false, bfr, (short)0, acc[i][j], false, false);
      }
    }
  }

  // ---- fused logit epilogue: -inv_var*(quad - 2*tn*S + tn^2*|gt|^2) ----
#pragma unroll
  for (int i = 0; i < 2; ++i) {
#pragma unroll
    for (int r = 0; r < 8; ++r) {
      int b = mTile + mWave * 32 + i * 16 + r + laneHi * 8;
      float tnb = tn[b], qb = quad[b], ivb = invvar[b];
#pragma unroll
      for (int j = 0; j < 4; ++j) {
        int p = nTile + nWave * 64 + j * 16 + lane15;
        if (p < kP) {
          float S  = acc[i][j][r];
          float lp = -ivb * (qb - 2.0f * tnb * S + tnb * tnb * sq[p]);
          logits[(size_t)b * kPpad + p] = lp;
        }
      }
    }
  }
}

// ---------------------------------------------------------------------------
// K4: row softmax over P logits; emit bf16 weights, zero-padded to Ppad.
// ---------------------------------------------------------------------------
__global__ __launch_bounds__(256) void softmax_kernel(
    const float* __restrict__ logits, __bf16* __restrict__ wts) {
  int b = blockIdx.x;
  const float* row = logits + (size_t)b * kPpad;
  __bf16* wrow = wts + (size_t)b * kPpad;
  __shared__ float red[256];

  float m = -3.4e38f;
  for (int p = threadIdx.x; p < kP; p += 256) m = fmaxf(m, row[p]);
  red[threadIdx.x] = m; __syncthreads();
  for (int off = 128; off; off >>= 1) {
    if ((int)threadIdx.x < off)
      red[threadIdx.x] = fmaxf(red[threadIdx.x], red[threadIdx.x + off]);
    __syncthreads();
  }
  float rmax = red[0]; __syncthreads();

  float s = 0.f;
  for (int p = threadIdx.x; p < kP; p += 256) s += __expf(row[p] - rmax);
  red[threadIdx.x] = s; __syncthreads();
  for (int off = 128; off; off >>= 1) {
    if ((int)threadIdx.x < off) red[threadIdx.x] += red[threadIdx.x + off];
    __syncthreads();
  }
  float inv = 1.0f / red[0];

  for (int p = threadIdx.x; p < kPpad; p += 256) {
    float w = (p < kP) ? __expf(row[p] - rmax) * inv : 0.0f;
    wrow[p] = (__bf16)w;
  }
}

// ---------------------------------------------------------------------------
// K5: wsum GEMM.  out = (1/(1-tn)) * (weights @ gt - xt).
// M=256 (all B in one workgroup -> gt streamed exactly once), N-tile=64,
// K over Ppad step 32.  gt tiles (32 x 64 f32) arrive via TDM double-buffered;
// an LDS->LDS transpose+convert phase builds the bf16 B tile; A fragments
// load straight from global bf16 weights (K-contiguous).
// ---------------------------------------------------------------------------
__global__ __launch_bounds__(256) void gemm2_kernel(
    const __bf16* __restrict__ wts, const float* __restrict__ gt,
    const float* __restrict__ xt, const float* __restrict__ inv1mt,
    float* __restrict__ out) {
  __shared__ float  Gf32[2][32 * PG];  // raw gt tile [p_local][d_local]
  __shared__ __bf16 BT[64 * PBT];      // transposed  [d_local][p_local]

  const int tid    = threadIdx.x;
  const int lane   = tid & 31;
  const int wave   = tid >> 5;
  const int lane15 = lane & 15;
  const int laneHi = lane >> 4;
  const int n0     = blockIdx.x * 64;  // column tile over D
  const int mWave  = wave;             // 0..7 -> 32 M rows each

  const unsigned ldsG[2] = { (unsigned)(uintptr_t)&Gf32[0][0],
                             (unsigned)(uintptr_t)&Gf32[1][0] };

  v8f acc[2][4] = {};
  constexpr int nSteps = kPpad / 32;   // 1564

  if (wave == 0) {
    tdm_load_2d_f32(ldsG[0], gt + n0, kP, kD, 64, 32, 5);
  }

  for (int ks = 0; ks < nSteps; ++ks) {
    const int buf = ks & 1;
    const int k0  = ks * 32;
    if (wave == 0) {
      if (ks + 1 < nSteps) {
        const int k0n = (ks + 1) * 32;
        tdm_load_2d_f32(ldsG[buf ^ 1], gt + (size_t)k0n * kD + n0,
                        kP - k0n, kD, 64, 32, 5);
        __builtin_amdgcn_s_wait_tensorcnt(1);   // tile for step ks is done
      } else {
        __builtin_amdgcn_s_wait_tensorcnt(0);
      }
    }
    __syncthreads();  // B1: f32 tile visible; prior compute done (BT reuse safe)

    // ---- transpose + convert: Gf32[pl][dl] -> BT[dl][pl] (bf16) ----
    {
      const float* Gs = Gf32[buf];
#pragma unroll
      for (int i = 0; i < 2; ++i) {
        int idx = tid + i * 256;          // 0..511 float4 slots
        int pl  = idx >> 4;               // 0..31  (K within tile)
        int c4  = idx & 15;               // 0..15  float4 column
        float4 v = *(const float4*)(Gs + pl * PG + c4 * 4);
        int dl = c4 * 4;
        BT[(dl + 0) * PBT + pl] = (__bf16)v.x;
        BT[(dl + 1) * PBT + pl] = (__bf16)v.y;
        BT[(dl + 2) * PBT + pl] = (__bf16)v.z;
        BT[(dl + 3) * PBT + pl] = (__bf16)v.w;
      }
    }
    __syncthreads();  // B2: BT ready

    v16bf a[2];
#pragma unroll
    for (int i = 0; i < 2; ++i) {
      int b = mWave * 32 + i * 16 + lane15;
      size_t base = (size_t)b * kPpad + k0 + (size_t)(laneHi * 8);
      a[i] = mk16(ld8(wts + base), ld8(wts + base + 16));
    }
#pragma unroll
    for (int j = 0; j < 4; ++j) {
      int d   = j * 16 + lane15;
      int ks2 = laneHi * 8;
      v16bf bfr = mk16(ld8(&BT[d * PBT + ks2]), ld8(&BT[d * PBT + ks2 + 16]));
#pragma unroll
      for (int i = 0; i < 2; ++i) {
        acc[i][j] = __builtin_amdgcn_wmma_f32_16x16x32_bf16(
            false, a[i], false, bfr, (short)0, acc[i][j], false, false);
      }
    }
  }

  // ---- fused final epilogue ----
#pragma unroll
  for (int i = 0; i < 2; ++i) {
#pragma unroll
    for (int r = 0; r < 8; ++r) {
      int b = mWave * 32 + i * 16 + r + laneHi * 8;
      float sc = inv1mt[b];
#pragma unroll
      for (int j = 0; j < 4; ++j) {
        int d = n0 + j * 16 + lane15;
        size_t o = (size_t)b * kD + d;
        out[o] = sc * (acc[i][j][r] - xt[o]);
      }
    }
  }
}

// ---------------------------------------------------------------------------
// Launcher
// ---------------------------------------------------------------------------
extern "C" void kernel_launch(void* const* d_in, const int* in_sizes, int n_in,
                              void* d_out, int out_size, void* d_ws, size_t ws_size,
                              hipStream_t stream) {
  (void)in_sizes; (void)n_in; (void)out_size; (void)ws_size;
  const float* xt = (const float*)d_in[0];   // [256,3,32,32]
  const float* t  = (const float*)d_in[1];   // [256]
  const float* gt = (const float*)d_in[2];   // [50000,3,32,32]
  float* out = (float*)d_out;                // [256,3,32,32]

  char* ws = (char*)d_ws;
  float*  tn      = (float*)(ws + 0);
  float*  quad    = (float*)(ws + 1024);
  float*  invvar  = (float*)(ws + 2048);
  float*  inv1mt  = (float*)(ws + 3072);
  float*  sq      = (float*)(ws + 4096);                 // P*4 = 200000 B
  float*  logits  = (float*)(ws + 204800);               // B*Ppad*4 = 51,249,152 B
  __bf16* wts     = (__bf16*)(ws + 204800 + (size_t)kB * kPpad * 4);  // B*Ppad*2

  prep_b_kernel<<<kB, 256, 0, stream>>>(xt, t, tn, quad, invvar, inv1mt);
  prep_p_kernel<<<kP, 256, 0, stream>>>(gt, sq);
  gemm1_kernel<<<dim3(kPpad / 128, kB / 128), 256, 0, stream>>>(
      xt, gt, tn, quad, invvar, sq, logits);
  softmax_kernel<<<kB, 256, 0, stream>>>(logits, wts);
  gemm2_kernel<<<kD / 64, 256, 0, stream>>>(wts, gt, xt, inv1mt, out);
}
